// MyNet_18279380811923
// MI455X (gfx1250) — matrix operand order repaired and checked by reading, not compiled, not run on hardware
//
#include <hip/hip_runtime.h>

// ---------------------------------------------------------------------------
// Types for CDNA5 WMMA (wave32): v_wmma_f32_16x16x32_bf16
// ---------------------------------------------------------------------------
typedef __attribute__((ext_vector_type(8)))  __bf16 v8bf;
typedef __attribute__((ext_vector_type(16))) __bf16 v16bf;
typedef __attribute__((ext_vector_type(8)))  float  v8f;

#define BN_EPS 1e-5f

#if defined(__gfx1250__) && __has_builtin(__builtin_amdgcn_global_load_async_to_lds_b128)
#define HAVE_ASYNC_LDS 1
typedef int v4i_vs __attribute__((vector_size(16)));
typedef __attribute__((address_space(1))) v4i_vs* gv4i_p;  // global
typedef __attribute__((address_space(3))) v4i_vs* lv4i_p;  // LDS
__device__ __forceinline__ void wait_async0() {
#if __has_builtin(__builtin_amdgcn_s_wait_asynccnt)
  __builtin_amdgcn_s_wait_asynccnt(0);
#else
  asm volatile("s_wait_asynccnt 0" ::: "memory");
#endif
}
#endif

// ---------------------------------------------------------------------------
// Weight conversion + reorder:  src OIHW [co][ci][r9] f32  ->
//   dst [co][r9*256 + ci] bf16  (so each K-chunk of 32 sits inside one r9)
// ---------------------------------------------------------------------------
__global__ void k_cvt_perm(const float* __restrict__ src, __bf16* __restrict__ dst, int n) {
  int i = blockIdx.x * 256 + threadIdx.x;
  if (i >= n) return;
  int co  = i / 2304;
  int rem = i - co * 2304;
  int ci  = rem / 9;
  int r9  = rem - ci * 9;
  dst[co * 2304 + r9 * 256 + ci] = (__bf16)src[i];
}

// ---------------------------------------------------------------------------
// conv1: 1 input channel, 4x4, stride 4, no pad; fused BN + ReLU + roll(256).
// ---------------------------------------------------------------------------
__global__ __launch_bounds__(256) void k_conv1_roll(
    const float* __restrict__ x, const float* __restrict__ w, const float* __restrict__ b,
    const float* __restrict__ g, const float* __restrict__ bb,
    const float* __restrict__ m, const float* __restrict__ vv,
    __bf16* __restrict__ out) {
  int pix = blockIdx.x;            // 0..65535
  int n   = blockIdx.y;            // 0..1
  int oy = pix >> 8, ox = pix & 255;
  __shared__ float patch[16];
  if (threadIdx.x < 16) {
    int k = threadIdx.x;
    int iy = oy * 4 + (k >> 2), ix = ox * 4 + (k & 3);
    patch[k] = x[((size_t)n * 1024 + iy) * 1024 + ix];
  }
  __syncthreads();
  int c = threadIdx.x;
  float acc = b[c];
#pragma unroll
  for (int k = 0; k < 16; ++k) acc += w[c * 16 + k] * patch[k];
  float sc = g[c] * rsqrtf(vv[c] + BN_EPS);
  acc = acc * sc + (bb[c] - m[c] * sc);
  acc = fmaxf(acc, 0.f);
  int ry = oy, rx = ox;
  if (c < 16) { ry = (oy + 64 * (c >> 2)) & 255; rx = (ox + 64 * (c & 3)) & 255; }
  out[(((size_t)n * 256 + c) * 256 + ry) * 256 + rx] = (__bf16)acc;
}

// ---------------------------------------------------------------------------
// Implicit-GEMM 3x3 conv (pad 1), 256->256 ch, bf16 in/out, f32 WMMA accum,
// fused bias+BN+ReLU.
//   Output tile per WG: 128 channels x 64 pixels; 8 wave32 waves (4 M x 2 N).
//   Each wave holds a 2x2 fragment grid (rows wm*16 / wm*16+64, cols wn*32 /
//   wn*32+16) -> 4 WMMAs per K-stage with register-level A/B reuse.
//   K = 9*256, stage = 32; weights pre-reordered to [co][r9*256+ci] so the
//   im2col decode is loop-invariant per stage (no div/mod in the hot loop).
//   LDS double-buffered -> single barrier per stage. A-tile staged with
//   global_load_async_to_lds_b128 (ASYNCcnt) when available.
// ---------------------------------------------------------------------------
__global__ __launch_bounds__(256) void k_conv3x3_wmma(
    const __bf16* __restrict__ in, const __bf16* __restrict__ wgt,
    const float* __restrict__ bias,
    const float* __restrict__ g, const float* __restrict__ bb,
    const float* __restrict__ mm, const float* __restrict__ vv,
    __bf16* __restrict__ out, int S_in, int S_out, int stride) {
  __shared__ __align__(64) __bf16 ldsA[2][128 * 32];  // [buf][row][k]  8 KB/buf
  __shared__ __align__(64) __bf16 ldsB[2][64 * 32];   // [buf][col][k]  4 KB/buf

  __builtin_amdgcn_s_wait_tensorcnt(0);

  const int t    = threadIdx.x;
  const int lane = t & 31;
  const int wid  = t >> 5;          // 0..7
  const int wm   = wid & 3;         // 4 M-subtiles (16 rows each, x2 with +64)
  const int wn   = wid >> 2;        // 2 N-subtiles (32 cols each)
  const int mt   = blockIdx.y;      // 0..1 -> 128 channels each
  const int p0   = blockIdx.x * 64; // pixel tile base (tiles always full)
  const int SPo  = S_out * S_out;
  const int SPi  = S_in * S_in;
  const int NSTAGE = 72;            // 9 * (256/32)

  // ---- B-gather role (fixed for whole kernel): col = t&63, 8 ci's at bks ----
  const int bcol = t & 63;
  const int bks  = (t >> 6) * 8;    // 0,8,16,24
  const int p    = p0 + bcol;
  const int bn_  = p / SPo;
  const int prem = p - bn_ * SPo;
  const int boy  = prem / S_out, box = prem - boy * S_out;
  const __bf16* inb = in + (size_t)bn_ * 256 * SPi;

  // ---- A-load role: row = t>>1 (0..127), 16 bf16 at (t&1)*16 ----
  const int arow = t >> 1;
  const int ac0  = (t & 1) * 16;
  const __bf16* wrow = wgt + ((size_t)(mt * 128 + arow)) * 2304 + ac0;
  const int aoff = arow * 32 + ac0;

  v8f acc00 = {}, acc01 = {}, acc10 = {}, acc11 = {};

  auto stage = [&](int s, int buf) {
    // A tile: 2 x 16B contiguous per lane -> async copy to LDS when available
    const __bf16* src = wrow + s * 32;
#ifdef HAVE_ASYNC_LDS
    __builtin_amdgcn_global_load_async_to_lds_b128(
        (gv4i_p)src, (lv4i_p)&ldsA[buf][aoff], 0, 0);
    __builtin_amdgcn_global_load_async_to_lds_b128(
        (gv4i_p)(src + 8), (lv4i_p)&ldsA[buf][aoff + 8], 0, 0);
#else
    *(v8bf*)&ldsA[buf][aoff]     = *(const v8bf*)(src);
    *(v8bf*)&ldsA[buf][aoff + 8] = *(const v8bf*)(src + 8);
#endif
    // B tile: im2col gather, branch-free (clamped address + select-zero)
    const int r9 = s >> 3;
    const int cibase = (s & 7) * 32;
    const int rr = (r9 * 11) >> 5;       // r9 / 3 for r9 in [0,8]
    const int ss = r9 - rr * 3;
    int iy = boy * stride - 1 + rr;
    int ix = box * stride - 1 + ss;
    bool valid = ((unsigned)iy < (unsigned)S_in) & ((unsigned)ix < (unsigned)S_in);
    int iyc = min(max(iy, 0), S_in - 1);
    int ixc = min(max(ix, 0), S_in - 1);
    const __bf16* bptr = inb + (size_t)iyc * S_in + ixc + (size_t)(cibase + bks) * SPi;
    v8bf bv;
#pragma unroll
    for (int q = 0; q < 8; ++q) {
      __bf16 val = bptr[(size_t)q * SPi];
      bv[q] = valid ? val : (__bf16)0.0f;
    }
    *(v8bf*)&ldsB[buf][bcol * 32 + bks] = bv;
  };

  auto loadA = [&](int buf, int row) -> v16bf {
    const int ko = (lane < 16) ? 0 : 16;
    v8bf a0 = *(const v8bf*)&ldsA[buf][row * 32 + ko];
    v8bf a1 = *(const v8bf*)&ldsA[buf][row * 32 + ko + 8];
    return __builtin_shufflevector(a0, a1, 0, 1, 2, 3, 4, 5, 6, 7,
                                   8, 9, 10, 11, 12, 13, 14, 15);
  };

  auto compute = [&](int buf) {
    const int row = wm * 16 + (lane & 15);
    v16bf a0 = loadA(buf, row);
    v16bf a1 = loadA(buf, row + 64);
    const int col = wn * 32 + (lane & 15);
    const int ks  = (lane < 16) ? 0 : 16;
    v16bf b0 = *(const v16bf*)&ldsB[buf][col * 32 + ks];
    v16bf b1 = *(const v16bf*)&ldsB[buf][(col + 16) * 32 + ks];
    acc00 = __builtin_amdgcn_wmma_f32_16x16x32_bf16(false, a0, false, b0, (short)0, acc00, false, false);
    acc01 = __builtin_amdgcn_wmma_f32_16x16x32_bf16(false, a0, false, b1, (short)0, acc01, false, false);
    acc10 = __builtin_amdgcn_wmma_f32_16x16x32_bf16(false, a1, false, b0, (short)0, acc10, false, false);
    acc11 = __builtin_amdgcn_wmma_f32_16x16x32_bf16(false, a1, false, b1, (short)0, acc11, false, false);
  };

  stage(0, 0);
#ifdef HAVE_ASYNC_LDS
  wait_async0();
#endif
  __syncthreads();
  for (int s = 0; s < NSTAGE; ++s) {
    if (s + 1 < NSTAGE) stage(s + 1, (s + 1) & 1);
    compute(s & 1);
#ifdef HAVE_ASYNC_LDS
    wait_async0();
#endif
    __syncthreads();
  }

  // ---- epilogue: C/D layout lane->N, vgpr v -> M=v (+8 for lanes 16-31) ----
#pragma unroll
  for (int mi = 0; mi < 2; ++mi) {
#pragma unroll
    for (int j = 0; j < 2; ++j) {
      const v8f& acc = mi ? (j ? acc11 : acc10) : (j ? acc01 : acc00);
      int col  = wn * 32 + j * 16 + (lane & 15);
      int pp   = p0 + col;
      int on   = pp / SPo;
      int orem = pp - on * SPo;
      int ooy  = orem / S_out, oox = orem - ooy * S_out;
#pragma unroll
      for (int v = 0; v < 8; ++v) {
        int m = mt * 128 + mi * 64 + wm * 16 + v + ((lane < 16) ? 0 : 8);
        float sc  = g[m] * rsqrtf(vv[m] + BN_EPS);
        float val = acc[v] + bias[m];
        val = val * sc + (bb[m] - mm[m] * sc);
        val = fmaxf(val, 0.f);
        out[(((size_t)on * 256 + m) * S_out + ooy) * S_out + oox] = (__bf16)val;
      }
    }
  }
}

// ---------------------------------------------------------------------------
// out = relu(convb + maxpool3x3s2p1(x)), fused with roll(S2) on the store.
// ---------------------------------------------------------------------------
__global__ __launch_bounds__(256) void k_pool_add_roll(
    const __bf16* __restrict__ xin, int S,
    const __bf16* __restrict__ conv, __bf16* __restrict__ out, int S2) {
  int e = blockIdx.x * 256 + threadIdx.x;
  int total = 2 * 256 * S2 * S2;
  if (e >= total) return;
  int ox = e % S2; int t = e / S2;
  int oy = t % S2; t /= S2;
  int c  = t % 256; int n = t / 256;
  size_t base = ((size_t)n * 256 + c) * S * S;
  float mx = -3.4e38f;
#pragma unroll
  for (int dy = -1; dy <= 1; ++dy) {
    int iy = oy * 2 + dy;
    if ((unsigned)iy >= (unsigned)S) continue;
#pragma unroll
    for (int dx = -1; dx <= 1; ++dx) {
      int ix = ox * 2 + dx;
      if ((unsigned)ix >= (unsigned)S) continue;
      mx = fmaxf(mx, (float)xin[base + (size_t)iy * S + ix]);
    }
  }
  float v = (float)conv[e] + mx;
  v = fmaxf(v, 0.f);
  int ts = S2 >> 2;
  int ry = oy, rx = ox;
  if (c < 16) { ry = (oy + ts * (c >> 2)) & (S2 - 1); rx = (ox + ts * (c & 3)) & (S2 - 1); }
  out[(((size_t)n * 256 + c) * S2 + ry) * S2 + rx] = (__bf16)v;
}

// ---------------------------------------------------------------------------
// adaptive avg pool 16x16 -> 1x1 : one block per (n,c)
// ---------------------------------------------------------------------------
__global__ __launch_bounds__(256) void k_avgpool(const __bf16* __restrict__ in,
                                                 float* __restrict__ pooled) {
  int nc = blockIdx.x;  // 0..511
  __shared__ float red[256];
  red[threadIdx.x] = (float)in[(size_t)nc * 256 + threadIdx.x];
  __syncthreads();
  for (int s = 128; s > 0; s >>= 1) {
    if (threadIdx.x < s) red[threadIdx.x] += red[threadIdx.x + s];
    __syncthreads();
  }
  if (threadIdx.x == 0) pooled[nc] = red[0] * (1.0f / 256.0f);
}

// ---------------------------------------------------------------------------
// fc1: h[n,i] = sum_j pooled[n,j] * W[i,j] + b[i]
// ---------------------------------------------------------------------------
__global__ __launch_bounds__(256) void k_fc1(const float* __restrict__ pooled,
                                             const float* __restrict__ w,
                                             const float* __restrict__ b,
                                             float* __restrict__ h) {
  int n = blockIdx.x, i = threadIdx.x;
  __shared__ float pin[256];
  pin[i] = pooled[n * 256 + i];
  __syncthreads();
  float acc = b[i];
  for (int j = 0; j < 256; ++j) acc += pin[j] * w[i * 256 + j];
  h[n * 256 + i] = acc;
}

// ---------------------------------------------------------------------------
// fc2: out[n,o] = sum_j h[n,j] * W[o,j] + b[o]
// ---------------------------------------------------------------------------
__global__ __launch_bounds__(256) void k_fc2(const float* __restrict__ h,
                                             const float* __restrict__ w,
                                             const float* __restrict__ b,
                                             float* __restrict__ out) {
  int idx = blockIdx.x * 256 + threadIdx.x;
  if (idx >= 2 * 12576) return;
  int n = idx / 12576, o = idx - n * 12576;
  const float* hr = h + n * 256;
  float acc = b[o];
  for (int j = 0; j < 256; ++j) acc += hr[j] * w[(size_t)o * 256 + j];
  out[idx] = acc;
}

// ---------------------------------------------------------------------------
// Launcher.  Input flattening (DFS over setup_inputs dict):
//  0:x 1:conv1_w 2:conv1_b 3-6:bn1 7:conv2_w 8:conv2_b 9-12:bn2
//  13:fc1_w 14:fc1_b 15:fc2_w 16:fc2_b
//  17-28: layer1, 29-40: layer2, 41-52: layer3
// ---------------------------------------------------------------------------
extern "C" void kernel_launch(void* const* d_in, const int* in_sizes, int n_in,
                              void* d_out, int out_size, void* d_ws, size_t ws_size,
                              hipStream_t stream) {
  (void)in_sizes; (void)n_in; (void)out_size; (void)ws_size;
  auto F = [&](int i) { return (const float*)d_in[i]; };

  const int WN = 256 * 256 * 9;  // 589824 weights per 3x3 conv

  char* ws = (char*)d_ws;
  size_t off = 0;
  auto carve = [&](size_t bytes) { void* p = ws + off; off += (bytes + 255) & ~(size_t)255; return p; };
  __bf16* w_l1c1 = (__bf16*)carve((size_t)WN * 2);
  __bf16* w_l1c2 = (__bf16*)carve((size_t)WN * 2);
  __bf16* w_l2c1 = (__bf16*)carve((size_t)WN * 2);
  __bf16* w_l2c2 = (__bf16*)carve((size_t)WN * 2);
  __bf16* w_l3c1 = (__bf16*)carve((size_t)WN * 2);
  __bf16* w_l3c2 = (__bf16*)carve((size_t)WN * 2);
  __bf16* w_fin  = (__bf16*)carve((size_t)WN * 2);
  __bf16* A0 = (__bf16*)carve((size_t)2 * 256 * 256 * 256 * 2);
  __bf16* T1 = (__bf16*)carve((size_t)2 * 256 * 128 * 128 * 2);
  __bf16* T2 = (__bf16*)carve((size_t)2 * 256 * 128 * 128 * 2);
  __bf16* A1 = (__bf16*)carve((size_t)2 * 256 * 128 * 128 * 2);
  __bf16* A2 = (__bf16*)carve((size_t)2 * 256 * 64 * 64 * 2);
  __bf16* A3 = (__bf16*)carve((size_t)2 * 256 * 32 * 32 * 2);
  __bf16* Ff = (__bf16*)carve((size_t)2 * 256 * 16 * 16 * 2);
  float* pooled = (float*)carve(512 * 4);
  float* h1     = (float*)carve(512 * 4);

  int cvtg = (WN + 255) / 256;
  k_cvt_perm<<<cvtg, 256, 0, stream>>>(F(17), w_l1c1, WN);
  k_cvt_perm<<<cvtg, 256, 0, stream>>>(F(23), w_l1c2, WN);
  k_cvt_perm<<<cvtg, 256, 0, stream>>>(F(29), w_l2c1, WN);
  k_cvt_perm<<<cvtg, 256, 0, stream>>>(F(35), w_l2c2, WN);
  k_cvt_perm<<<cvtg, 256, 0, stream>>>(F(41), w_l3c1, WN);
  k_cvt_perm<<<cvtg, 256, 0, stream>>>(F(47), w_l3c2, WN);
  k_cvt_perm<<<cvtg, 256, 0, stream>>>(F(7),  w_fin,  WN);

  k_conv1_roll<<<dim3(65536, 2), 256, 0, stream>>>(F(0), F(1), F(2), F(3), F(4), F(5), F(6), A0);

  auto conv = [&](const __bf16* in, const __bf16* w, int bi, int bni,
                  __bf16* out, int Sin, int Sout, int stride) {
    int ntiles = (2 * Sout * Sout) / 64;
    k_conv3x3_wmma<<<dim3(ntiles, 2), 256, 0, stream>>>(
        in, w, F(bi), F(bni), F(bni + 1), F(bni + 2), F(bni + 3), out, Sin, Sout, stride);
  };
  auto pool = [&](const __bf16* x, int S, const __bf16* cv, __bf16* out) {
    int S2 = S / 2;
    int total = 2 * 256 * S2 * S2;
    k_pool_add_roll<<<(total + 255) / 256, 256, 0, stream>>>(x, S, cv, out, S2);
  };

  conv(A0, w_l1c1, 18, 19, T1, 256, 128, 2);
  conv(T1, w_l1c2, 24, 25, T2, 128, 128, 1);
  pool(A0, 256, T2, A1);
  conv(A1, w_l2c1, 30, 31, T1, 128, 64, 2);
  conv(T1, w_l2c2, 36, 37, T2, 64, 64, 1);
  pool(A1, 128, T2, A2);
  conv(A2, w_l3c1, 42, 43, T1, 64, 32, 2);
  conv(T1, w_l3c2, 48, 49, T2, 32, 32, 1);
  pool(A2, 64, T2, A3);
  conv(A3, w_fin, 8, 9, Ff, 32, 16, 2);

  k_avgpool<<<512, 256, 0, stream>>>(Ff, pooled);
  k_fc1<<<2, 256, 0, stream>>>(pooled, F(13), F(14), h1);
  k_fc2<<<(2 * 12576 + 255) / 256, 256, 0, stream>>>(h1, F(15), F(16), (float*)d_out);
}
